// IndRNNNet_25357486916094
// MI455X (gfx1250) — matrix operand order repaired and checked by reading, not compiled, not run on hardware
//
#include <hip/hip_runtime.h>

// ---------------------------------------------------------------------------
// IndRNN net: proj GEMM (WMMA f16) -> diagonal scan -> small classifier
//   x     [64,512] i32      emb  [32000,256] f32
//   W_in  [1024,256] f32    b_in [1024] f32
//   u     [1024] f32        b_cell [1024] f32
//   W_fc  [10,1024] f32     b_fc [10] f32
//   out   [64,10] f32
// ---------------------------------------------------------------------------

typedef _Float16 v16h __attribute__((ext_vector_type(16)));
typedef _Float16 v8h  __attribute__((ext_vector_type(8)));
typedef _Float16 v4h  __attribute__((ext_vector_type(4)));
typedef float    v8f  __attribute__((ext_vector_type(8)));

#define BATCH 64
#define SEQ   512
#define EMBD  256
#define HID   1024
#define NCLS  10
#define MTOT  (BATCH * SEQ)          // 32768 rows of the projection GEMM

// workspace layout (bytes)
#define BSW_OFF   0                          // swizzled W_in fragments, 512 KB
#define PROJ_OFF  (1u << 20)                 // proj f16, 64 MB
#define HFIN_OFF  (PROJ_OFF + (size_t)MTOT * HID * sizeof(_Float16))

#define ASTRIDE 264   // LDS row stride in halves (256 + 8 pad)

// ---------------------------------------------------------------------------
// Phase 0: swizzle W_in (f32, [H][E]) into per-lane WMMA B fragments (f16).
// Fragment block (kb in 0..7, nb in 0..63): lane l holds 16 contiguous halves
//   B[k][n] = W_in[n][k] with n = nb*16 + (l&15), k = kb*32 + (l>>4)*16 + e.
// ---------------------------------------------------------------------------
__global__ void indrnn_swizzle_b(const float* __restrict__ Win,
                                 _Float16* __restrict__ Bsw) {
    int t = blockIdx.x * blockDim.x + threadIdx.x;   // 16384 threads
    int lane = t & 31;
    int nb   = (t >> 5) & 63;
    int kb   = t >> 11;
    int n = nb * 16 + (lane & 15);
    int k = kb * 32 + (lane >> 4) * 16;
    const float* src = Win + (size_t)n * EMBD + k;   // 16 contiguous f32
    _Float16* dst = Bsw + (size_t)t * 16;            // 32 contiguous bytes
#pragma unroll
    for (int e = 0; e < 16; ++e) dst[e] = (_Float16)src[e];
}

// ---------------------------------------------------------------------------
// Phase 1: proj[m][n] = sum_k emb[x[m]][k] * W_in[n][k] + (b_in+b_cell)[n]
// One workgroup = 4 waves = 16(M) x 256(N) tile, K loop of 8 (32 WMMAs).
// B fragments read off a single per-lane base with immediate offsets.
// ---------------------------------------------------------------------------
__global__ __launch_bounds__(128)
void indrnn_proj_gemm(const int*      __restrict__ x,
                      const float*    __restrict__ emb,
                      const _Float16* __restrict__ Bsw,
                      const float*    __restrict__ b_in,
                      const float*    __restrict__ b_cell,
                      _Float16*       __restrict__ proj) {
    __shared__ __align__(16) _Float16 sA[16 * ASTRIDE];

    const int m0  = blockIdx.x * 16;   // 0..2047 -> rows
    const int nc  = blockIdx.y;        // 0..3    -> 256-wide N chunk
    const int tid = threadIdx.x;

    // ---- stage A: gather 16 embedding rows, f32 -> f16, into LDS ----------
    for (int i = tid; i < 16 * 64; i += 128) {
        int r  = i >> 6;
        int c4 = (i & 63) * 4;
        int tok = x[m0 + r];
        const float* e4 = emb + (size_t)tok * EMBD + c4;
        v4h h4;
        h4[0] = (_Float16)e4[0];
        h4[1] = (_Float16)e4[1];
        h4[2] = (_Float16)e4[2];
        h4[3] = (_Float16)e4[3];
        *(v4h*)&sA[r * ASTRIDE + c4] = h4;
    }
    __syncthreads();

    const int wave = tid >> 5;
    const int lane = tid & 31;
    const int half = lane >> 4;
    const int l16  = lane & 15;

    const int nbase  = nc * 256 + wave * 64;     // global n of this wave's tile 0
    const int nbBase = nc * 16 + wave * 4;       // fragment-block index (n/16)

    // per-lane base: fragment (kb=0, nb=nbBase), this lane. All further
    // fragments are at compile-time byte offsets: kb*65536 + j*1024.
    const _Float16* bbase = Bsw + (size_t)(nbBase * 32 + lane) * 16;

    v8f acc[4];
#pragma unroll
    for (int j = 0; j < 4; ++j)
#pragma unroll
        for (int r = 0; r < 8; ++r) acc[j][r] = 0.0f;

#pragma unroll
    for (int kb = 0; kb < 8; ++kb) {
        const int k0 = kb * 32;
        // A fragment: rows m=l16, 16 contiguous halves at k0 + half*16
        v8h lo = *(const v8h*)&sA[l16 * ASTRIDE + k0 + half * 16];
        v8h hi = *(const v8h*)&sA[l16 * ASTRIDE + k0 + half * 16 + 8];
        v16h a = __builtin_shufflevector(lo, hi,
                                         0, 1, 2, 3, 4, 5, 6, 7,
                                         8, 9, 10, 11, 12, 13, 14, 15);
#pragma unroll
        for (int j = 0; j < 4; ++j) {
            v16h b = *(const v16h*)(bbase + kb * (64 * 32 * 16) + j * (32 * 16));
            acc[j] = __builtin_amdgcn_wmma_f32_16x16x32_f16(
                false, a, false, b, (short)0, acc[j], false, false);
        }
    }

    // ---- epilogue: fold bias, store f16 proj ------------------------------
    // per-lane store base; offsets r*HID + j*16 are constants after unroll.
    _Float16* pbase = proj + (size_t)(m0 + 8 * half) * HID + nbase + l16;
#pragma unroll
    for (int j = 0; j < 4; ++j) {
        int n = nbase + j * 16 + l16;
        float bias = b_in[n] + b_cell[n];
#pragma unroll
        for (int r = 0; r < 8; ++r)
            pbase[r * HID + j * 16] = (_Float16)(acc[j][r] + bias);
    }
}

// ---------------------------------------------------------------------------
// Phase 2: diagonal scan. Thread handles 8 consecutive h for one batch b.
//   h = relu(proj[b][t][h] + u[h]*h), t = 0..511.  Streaming b128 loads.
// ---------------------------------------------------------------------------
__global__ __launch_bounds__(128)
void indrnn_scan(const _Float16* __restrict__ proj,
                 const float*    __restrict__ u,
                 float*          __restrict__ hfin) {
    int g  = blockIdx.x * blockDim.x + threadIdx.x;  // 16384 threads
    int b  = g >> 7;
    int h0 = (g & 127) * 8;

    float uu[8], hh[8];
#pragma unroll
    for (int i = 0; i < 8; ++i) { uu[i] = u[h0 + i]; hh[i] = 0.0f; }

    const _Float16* p = proj + (size_t)b * SEQ * HID + h0;
#pragma unroll 4
    for (int t = 0; t < SEQ; ++t) {
        v8h v = *(const v8h*)p;
        p += HID;
#pragma unroll
        for (int i = 0; i < 8; ++i)
            hh[i] = fmaxf(0.0f, fmaf(uu[i], hh[i], (float)v[i]));
    }
#pragma unroll
    for (int i = 0; i < 8; ++i) hfin[(size_t)b * HID + h0 + i] = hh[i];
}

// ---------------------------------------------------------------------------
// Phase 3: classifier  out[b][c] = hfin[b] . W_fc[c] + b_fc[c]
// One block per batch element; wave c reduces class c over H=1024.
// ---------------------------------------------------------------------------
__global__ __launch_bounds__(320)
void indrnn_fc(const float* __restrict__ hfin,
               const float* __restrict__ Wfc,
               const float* __restrict__ bfc,
               float*       __restrict__ out) {
    int b = blockIdx.x;
    int c = threadIdx.x >> 5;   // 10 waves -> 10 classes
    int l = threadIdx.x & 31;

    const float* hv = hfin + (size_t)b * HID;
    const float* wv = Wfc + (size_t)c * HID;
    float sum = 0.0f;
    for (int h = l; h < HID; h += 32)
        sum = fmaf(hv[h], wv[h], sum);
#pragma unroll
    for (int off = 16; off > 0; off >>= 1)
        sum += __shfl_down(sum, off, 32);
    if (l == 0) out[b * NCLS + c] = sum + bfc[c];
}

// ---------------------------------------------------------------------------
extern "C" void kernel_launch(void* const* d_in, const int* in_sizes, int n_in,
                              void* d_out, int out_size, void* d_ws, size_t ws_size,
                              hipStream_t stream) {
    const int*   x      = (const int*)d_in[0];
    const float* emb    = (const float*)d_in[1];
    const float* W_in   = (const float*)d_in[2];
    const float* b_in   = (const float*)d_in[3];
    const float* u      = (const float*)d_in[4];
    const float* b_cell = (const float*)d_in[5];
    const float* W_fc   = (const float*)d_in[6];
    const float* b_fc   = (const float*)d_in[7];
    float* out = (float*)d_out;

    char* ws = (char*)d_ws;
    _Float16* Bsw  = (_Float16*)(ws + BSW_OFF);
    _Float16* proj = (_Float16*)(ws + PROJ_OFF);
    float*    hfin = (float*)(ws + HFIN_OFF);

    // Phase 0: swizzle W_in into WMMA B-fragment layout (f16)
    indrnn_swizzle_b<<<64, 256, 0, stream>>>(W_in, Bsw);

    // Phase 1: projection GEMM via v_wmma_f32_16x16x32_f16
    indrnn_proj_gemm<<<dim3(MTOT / 16, 4), 128, 0, stream>>>(
        x, emb, Bsw, b_in, b_cell, proj);

    // Phase 2: IndRNN diagonal scan
    indrnn_scan<<<128, 128, 0, stream>>>(proj, u, hfin);

    // Phase 3: classifier
    indrnn_fc<<<BATCH, 320, 0, stream>>>(hfin, W_fc, b_fc, out);
}
